// RNN_71665824301223
// MI455X (gfx1250) — compile-verified, matrix operand order checked by reading, not run on hardware
//
#include <hip/hip_runtime.h>
#include <math.h>

// Problem sizes (match reference)
#define BB 512
#define TT 512
#define II 256   // input dim (K for xproj GEMM)
#define HH 250   // hidden dim (valid N / valid K for recurrence)

typedef __attribute__((ext_vector_type(16))) __bf16        v16bf;
typedef __attribute__((ext_vector_type(8)))  float         v8f;
typedef __attribute__((ext_vector_type(8)))  unsigned int  u8u;

static __device__ __forceinline__ v16bf as_v16bf(u8u x) {
    return __builtin_bit_cast(v16bf, x);
}
static __device__ __forceinline__ u8u cat8(uint4 a, uint4 b) {
    u8u r;
    r[0] = a.x; r[1] = a.y; r[2] = a.z; r[3] = a.w;
    r[4] = b.x; r[5] = b.y; r[6] = b.z; r[7] = b.w;
    return r;
}
// Native f32 -> bf16 (RNE) via hardware cvt, packed into a dword
static __device__ __forceinline__ unsigned short bf16_u16(float f) {
    return __builtin_bit_cast(unsigned short, (__bf16)f);
}
static __device__ __forceinline__ unsigned pack2bf(float lo, float hi) {
    return (unsigned)bf16_u16(lo) | ((unsigned)bf16_u16(hi) << 16);
}

// CDNA5 async global->LDS copy (16B per lane), tracked by ASYNCcnt
static __device__ __forceinline__ void async_copy16(void* lds, const void* gmem) {
    unsigned           l = (unsigned)(unsigned long long)lds;  // low 32 bits = LDS offset
    unsigned long long g = (unsigned long long)gmem;
    asm volatile("global_load_async_to_lds_b128 %0, %1, off"
                 :: "v"(l), "v"(g) : "memory");
}
static __device__ __forceinline__ void async_wait0() {
    asm volatile("s_wait_asynccnt 0x0" ::: "memory");
}

// ---------------------------------------------------------------------------
// Fragment layout (CDNA5 ISA 7.12.2, 16-bit operands, wave32):
//  A (16x32 tile): lane l: row = l&15; VGPR v holds k = kt*32 + (v>=4?16:0)
//                  + (l>>4)*8 + (v&3)*2 (+half)  => for row-major storage the
//                  fragment is two contiguous 16B runs per lane.
//  B (32x16 tile): lane l: col = l&15; same k formula (symmetric layout).
//  C/D f32 16x16:  lane l: col = l&15; row = (l>>4)*8 + v.
// Pre-swizzled B fragments stored as dwords:
//  frag[kt*4096 + nt*256 + lane*8 + v]   (kt:8, nt:16, lane:32, v:8)
// ---------------------------------------------------------------------------

// Phase 1: Wx ([H,I]) and Wh ([H,H]) f32 -> padded bf16 B-matrix fragments.
// B[k][n] = W[n][k]; zero outside valid range (n>=250, and k>=250 for Wh).
__global__ __launch_bounds__(256)
void rnn_prep_weights(const float* __restrict__ Wx,
                      const float* __restrict__ Wh,
                      unsigned* __restrict__ wxfrag,
                      unsigned* __restrict__ whfrag) {
    int d = blockIdx.x * blockDim.x + threadIdx.x;   // 0..65535
    int which = d >> 15;                             // 0 = Wx, 1 = Wh
    int i = d & 32767;
    int v    = i & 7;
    int lane = (i >> 3) & 31;
    int nt   = (i >> 8) & 15;
    int kt   = (i >> 12) & 7;
    int n    = nt * 16 + (lane & 15);
    int hi   = lane >> 4;
    int k0   = kt * 32 + ((v & 4) ? 16 : 0) + hi * 8 + (v & 3) * 2;
    float a = 0.f, b = 0.f;
    if (which == 0) {
        if (n < HH) { a = Wx[n * II + k0]; b = Wx[n * II + k0 + 1]; }
        wxfrag[i] = pack2bf(a, b);
    } else {
        if (n < HH) {
            if (k0 < HH)     a = Wh[n * HH + k0];
            if (k0 + 1 < HH) b = Wh[n * HH + k0 + 1];
        }
        whfrag[i] = pack2bf(a, b);
    }
}

// Phase 2: xproj[b,t,n] = sum_k x[b,t,k]*Wx[n,k], written into d_out[b][t][n].
// Per WG: 64 rows (r = b*T+t), all 256 cols.  Wx fragments (128KB) and the
// bf16 x tile (64x264) both live in LDS; inner loop is DS-only.
#define XROW 264                         // padded bf16 row stride (bank spread)
#define XP_LDS (131072 + 64 * XROW * 2)  // 164864 B

__global__ __launch_bounds__(256)
void rnn_xproj(const float* __restrict__ x,
               const unsigned* __restrict__ wxfrag,
               float* __restrict__ out) {
    extern __shared__ char smem[];
    unsigned*       wxl = (unsigned*)smem;                     // 32768 dwords
    unsigned short* xl  = (unsigned short*)(smem + 131072);    // [64][264] bf16

    int tid  = threadIdx.x;
    int lane = tid & 31;
    int wave = tid >> 5;
    int lr   = lane & 15, hi = lane >> 4;
    int mt   = wave & 3;
    int ntb  = (wave >> 2) * 8;
    long rbase = (long)blockIdx.x * 64;

    // Async-stage Wx fragments (8192 x 16B) straight into LDS
    for (int i = tid; i < 8192; i += 256)
        async_copy16(wxl + i * 4, wxfrag + i * 4);
    // Stage x tile as bf16 (coalesced float4 reads, packed dword LDS stores)
    for (int idx = tid; idx < 64 * 64; idx += 256) {
        int r = idx >> 6, kq = idx & 63;
        float4 f = *(const float4*)(x + (rbase + r) * II + kq * 4);
        unsigned* dst = (unsigned*)xl + r * (XROW / 2) + kq * 2;
        dst[0] = pack2bf(f.x, f.y);
        dst[1] = pack2bf(f.z, f.w);
    }
    async_wait0();
    __syncthreads();

    v8f acc[8] = {};
    const unsigned short* arow = xl + (mt * 16 + lr) * XROW;

    for (int kt = 0; kt < 8; ++kt) {
        // A fragment: two contiguous 16B runs per lane (row-major layout)
        uint4 a0 = *(const uint4*)(arow + kt * 32 + hi * 8);
        uint4 a1 = *(const uint4*)(arow + kt * 32 + 16 + hi * 8);
        v16bf av = as_v16bf(cat8(a0, a1));
        for (int j = 0; j < 8; ++j) {
            const uint4* bp = (const uint4*)(wxl + ((kt * 16 + ntb + j) * 32 + lane) * 8);
            uint4 b0 = bp[0], b1 = bp[1];
            acc[j] = __builtin_amdgcn_wmma_f32_16x16x32_bf16(
                false, av, false, as_v16bf(cat8(b0, b1)), (short)0, acc[j], false, false);
        }
    }
    for (int j = 0; j < 8; ++j) {
        int n = (ntb + j) * 16 + lr;
        if (n < HH) {
            for (int v = 0; v < 8; ++v) {
                long m = rbase + mt * 16 + hi * 8 + v;    // D row
                out[m * HH + n] = acc[j][v];
            }
        }
    }
}

// Phase 3: recurrence.  One WG per 16 batch rows, looping t = 0..511.
// Wh fragments (128KB) async-staged to LDS once; h kept as padded bf16 tile.
#define PH_STRIDE 264
#define RC_LDS (131072 + 16 * PH_STRIDE * 2)   // 139520 B

__global__ __launch_bounds__(256)
void rnn_recurrence(const float* __restrict__ h0,
                    const unsigned* __restrict__ whfrag,
                    const float* __restrict__ bh,
                    float* __restrict__ out) {
    extern __shared__ char smem[];
    unsigned*       whl    = (unsigned*)smem;                    // 32768 dwords
    unsigned short* plainH = (unsigned short*)(smem + 131072);   // [16][264] bf16

    int tid  = threadIdx.x;
    int lane = tid & 31;
    int wave = tid >> 5;
    int lr   = lane & 15, hi = lane >> 4;
    int b0   = blockIdx.x * 16;

    for (int i = tid; i < 8192; i += 256)
        async_copy16(whl + i * 4, whfrag + i * 4);
    for (int i = tid; i < 16 * 256; i += 256) {
        int r = i >> 8, k = i & 255;
        float v = (k < HH) ? h0[(b0 + r) * HH + k] : 0.f;
        plainH[r * PH_STRIDE + k] = bf16_u16(v);
    }
    async_wait0();
    __syncthreads();

    int nt0 = wave * 2;
    int n0  = nt0 * 16 + lr;
    int n1  = n0 + 16;
    float bias0 = (n0 < HH) ? bh[n0] : 0.f;
    float bias1 = (n1 < HH) ? bh[n1] : 0.f;
    const unsigned short* arow = plainH + lr * PH_STRIDE;

    for (int t = 0; t < TT; ++t) {
        // acc starts as the xp_t tile (read back from d_out, D-fragment layout)
        v8f acc0, acc1;
        for (int v = 0; v < 8; ++v) {
            long m    = b0 + hi * 8 + v;
            long base = (m * TT + t) * (long)HH;
            acc0[v] = (n0 < HH) ? out[base + n0] : 0.f;
            acc1[v] = (n1 < HH) ? out[base + n1] : 0.f;
        }
        // acc += h_{t-1} @ Wh^T : A frag = two b128 LDS reads per lane
        for (int kt = 0; kt < 8; ++kt) {
            uint4 a0 = *(const uint4*)(arow + kt * 32 + hi * 8);
            uint4 a1 = *(const uint4*)(arow + kt * 32 + 16 + hi * 8);
            v16bf av = as_v16bf(cat8(a0, a1));
            {
                const uint4* bp = (const uint4*)(whl + ((kt * 16 + nt0) * 32 + lane) * 8);
                uint4 c0 = bp[0], c1 = bp[1];
                acc0 = __builtin_amdgcn_wmma_f32_16x16x32_bf16(
                    false, av, false, as_v16bf(cat8(c0, c1)), (short)0, acc0, false, false);
            }
            {
                const uint4* bp = (const uint4*)(whl + ((kt * 16 + nt0 + 1) * 32 + lane) * 8);
                uint4 c0 = bp[0], c1 = bp[1];
                acc1 = __builtin_amdgcn_wmma_f32_16x16x32_bf16(
                    false, av, false, as_v16bf(cat8(c0, c1)), (short)0, acc1, false, false);
            }
        }
        __syncthreads();   // all waves done reading h_{t-1}
        for (int v = 0; v < 8; ++v) {
            int  m  = hi * 8 + v;
            long gb = ((long)(b0 + m) * TT + t) * HH;
            float hv0 = tanhf(acc0[v] + bias0);
            float hv1 = tanhf(acc1[v] + bias1);
            if (n0 < HH) out[gb + n0] = hv0;
            if (n1 < HH) out[gb + n1] = hv1;
            plainH[m * PH_STRIDE + n0] = bf16_u16(hv0);
            plainH[m * PH_STRIDE + n1] = bf16_u16(hv1);
        }
        __syncthreads();   // h_t fully published before next step reads it
    }
}

extern "C" void kernel_launch(void* const* d_in, const int* in_sizes, int n_in,
                              void* d_out, int out_size, void* d_ws, size_t ws_size,
                              hipStream_t stream) {
    const float* x  = (const float*)d_in[0];   // [B,T,I]
    const float* h0 = (const float*)d_in[1];   // [B,H]
    const float* Wx = (const float*)d_in[2];   // [H,I]
    const float* Wh = (const float*)d_in[3];   // [H,H]
    const float* bh = (const float*)d_in[4];   // [H]
    float* out = (float*)d_out;                // [B,T,H]

    unsigned* wxfrag = (unsigned*)d_ws;        // 128 KB
    unsigned* whfrag = wxfrag + 32768;         // 128 KB

    rnn_prep_weights<<<256, 256, 0, stream>>>(Wx, Wh, wxfrag, whfrag);
    rnn_xproj<<<(BB * TT) / 64, 256, XP_LDS, stream>>>(x, wxfrag, out);
    rnn_recurrence<<<BB / 16, 256, RC_LDS, stream>>>(h0, whfrag, bh, out);
}
